// SogCLR_Penalty_22016002360043
// MI455X (gfx1250) — matrix operand order repaired and checked by reading, not compiled, not run on hardware
//
#include <hip/hip_runtime.h>

#define TEMP    0.07f
#define GAMMA1  0.8f
#define GAMMA2  0.8f
#define TAU_C   0.1f
#define BETA_C  40.0f
#define NUM_CT  5
#define EPS_F   1.1920928955078125e-07f

typedef __attribute__((ext_vector_type(16))) __bf16 v16bf;
typedef __attribute__((ext_vector_type(8)))  __bf16 v8bf;
typedef __attribute__((ext_vector_type(8)))  float  v8f;

typedef __attribute__((ext_vector_type(4))) unsigned int u32x4;
typedef __attribute__((ext_vector_type(8))) int          i32x8;
typedef __attribute__((ext_vector_type(4))) int          i32x4;

// LDS tile geometry: 32-k stage, 8-element (16B) row pad inserted by the TDM
// itself (pad_interval=3 -> every 16 DWORDs, pad_amount=3 -> 4 DWORDs).
#define KSTEP      32
#define LDS_STRIDE 40   // 32 + 8 pad elements; 80B row stride = 20 banks

// ---------------------------------------------------------------------------
// Issue one TDM tensor_load_to_lds: 2D bf16 tile (tile_d0 x tile_d1),
// row-major source with element stride `stride0`, into LDS at lds_addr.
// D# built per CDNA5 ISA 8.3/8.4 (group0: count/lds/global/type,
// group1: data_size/pad/dims/strides). Groups 2/3 zero (<=2D tensor).
// ---------------------------------------------------------------------------
__device__ __forceinline__
void tdm_load_tile(unsigned lds_addr, const __bf16* gptr,
                   int tile_d0, int tile_d1, unsigned long long stride0)
{
    unsigned long long ga = (unsigned long long)(size_t)gptr;

    u32x4 g0;
    g0[0] = 1u;                                   // count=1, user descriptor
    g0[1] = lds_addr;                             // LDS byte address
    g0[2] = (unsigned)(ga & 0xffffffffu);         // global_addr[31:0]
    g0[3] = (unsigned)((ga >> 32) & 0x1ffffffu)   // global_addr[56:32]
          | 0x80000000u;                          // type=2 ("image")

    const int dim_big = 0x7fffffff;               // no OOB clamping needed
    i32x8 g1;
    g1[0] = (1 << 16)                             // data_size = 2 bytes
          | (1 << 20)                             // pad_enable
          | (3 << 22)                             // pad_interval: 16 DWORDs
          | (3 << 25);                            // pad_amount:   4 DWORDs
    g1[1] = (dim_big & 0xffff) << 16;                             // tensor_dim0[15:0]
    g1[2] = ((dim_big >> 16) & 0xffff) | ((dim_big & 0xffff) << 16); // d0 hi | d1 lo
    g1[3] = ((dim_big >> 16) & 0xffff) | (tile_d0 << 16);         // d1 hi | tile_dim0
    g1[4] = tile_d1 & 0xffff;                                     // tile_dim1 (tile_dim2=0)
    g1[5] = (int)(stride0 & 0xffffffffu);                         // tensor_dim0_stride lo
    g1[6] = (int)((stride0 >> 32) & 0xffffu);                     // stride hi | d1_stride lo
    g1[7] = 0;

    i32x4 gz = {0, 0, 0, 0};
#if defined(__clang_major__) && (__clang_major__ >= 23)
    i32x8 gz8 = {0, 0, 0, 0, 0, 0, 0, 0};
    __builtin_amdgcn_tensor_load_to_lds(g0, g1, gz, gz, gz8, 0);
#else
    __builtin_amdgcn_tensor_load_to_lds(g0, g1, gz, gz, 0);
#endif
}

// ---------------------------------------------------------------------------
// Zero a float region (also zeroes the int counters living in the same span)
// ---------------------------------------------------------------------------
__global__ void zero_ws(float* __restrict__ p, int n) {
    int i = blockIdx.x * blockDim.x + threadIdx.x;
    if (i < n) p[i] = 0.0f;
}

// ---------------------------------------------------------------------------
// Convert f32 features -> bf16, compute diag[i] = dot(img[i], txt[i]) in f32,
// and count positives/negatives. One block (256 threads) per row.
// ---------------------------------------------------------------------------
__global__ __launch_bounds__(256)
void convert_diag(const float* __restrict__ img, const float* __restrict__ txt,
                  __bf16* __restrict__ imgB, __bf16* __restrict__ txtB,
                  float* __restrict__ diag, const int* __restrict__ slabel,
                  int* __restrict__ cnts, int D)
{
    int row = blockIdx.x;
    int t   = threadIdx.x;
    const float* ip = img + (size_t)row * D;
    const float* tp = txt + (size_t)row * D;
    __bf16* ib = imgB + (size_t)row * D;
    __bf16* tb = txtB + (size_t)row * D;
    float p = 0.0f;
    for (int e = t; e < D; e += 256) {
        float a = ip[e], b = tp[e];
        ib[e] = (__bf16)a;
        tb[e] = (__bf16)b;
        p += a * b;
    }
#pragma unroll
    for (int msk = 16; msk >= 1; msk >>= 1) p += __shfl_xor(p, msk, 32);
    __shared__ float red[8];
    if ((t & 31) == 0) red[t >> 5] = p;
    __syncthreads();
    if (t == 0) {
        float s = 0.0f;
#pragma unroll
        for (int i = 0; i < 8; ++i) s += red[i];
        diag[row] = s;
        atomicAdd(&cnts[(slabel[row] == 1) ? 0 : 1], 1);
    }
}

// ---------------------------------------------------------------------------
// Fused bf16 WMMA GEMM (sim tile) + SogCLR epilogue, TDM-staged through LDS.
// Block = 256 threads = 8 waves (wave32), arranged 4 (M) x 2 (N).
// Block tile 128x128; wave tile 32x64 = 2x4 tiles of 16x16; K staged 32/step.
// Wave 0 drives a double-buffered tensor_load_to_lds pipeline (TENSORcnt).
// ---------------------------------------------------------------------------
__global__ __launch_bounds__(256)
void gemm_epilogue(const __bf16* __restrict__ imgB, const __bf16* __restrict__ txtB,
                   const float* __restrict__ diag, const int* __restrict__ slabel,
                   float* __restrict__ Arow, float* __restrict__ Brow,
                   float* __restrict__ Acol, float* __restrict__ Bcol,
                   int D)
{
    const float invT = 1.0f / TEMP;
    __shared__ __bf16 sA[2][128 * LDS_STRIDE];
    __shared__ __bf16 sB[2][128 * LDS_STRIDE];
    __shared__ float s_drow[128], s_dcol[128], s_erow[128], s_ecol[128];
    __shared__ int   s_lrow[128], s_lcol[128];

    const int t        = threadIdx.x;
    const int row_base = blockIdx.y * 128;
    const int col_base = blockIdx.x * 128;

    if (t < 128) {
        float dv = diag[row_base + t];
        s_drow[t] = dv;
        s_erow[t] = __expf(-dv * invT);
        s_lrow[t] = slabel[row_base + t];
    } else {
        int u2 = t - 128;
        float dv = diag[col_base + u2];
        s_dcol[u2] = dv;
        s_ecol[u2] = __expf(-dv * invT);
        s_lcol[u2] = slabel[col_base + u2];
    }

    const int wave = t >> 5, lane = t & 31;
    const int wm = wave & 3, wn = wave >> 2;   // 4 x 2 wave grid
    const int half = lane >> 4, lrow = lane & 15;

    v8f acc[2][4];
#pragma unroll
    for (int i = 0; i < 2; ++i)
#pragma unroll
        for (int j = 0; j < 4; ++j) {
            v8f z = {0.f, 0.f, 0.f, 0.f, 0.f, 0.f, 0.f, 0.f};
            acc[i][j] = z;
        }

    const __bf16* gA = imgB + (size_t)row_base * D;
    const __bf16* gB = txtB + (size_t)col_base * D;
    unsigned ldsA0 = (unsigned)(size_t)&sA[0][0];
    unsigned ldsA1 = (unsigned)(size_t)&sA[1][0];
    unsigned ldsB0 = (unsigned)(size_t)&sB[0][0];
    unsigned ldsB1 = (unsigned)(size_t)&sB[1][0];

    const int nstage = D / KSTEP;
    if (wave == 0) {                       // preload stage 0
        tdm_load_tile(ldsA0, gA, KSTEP, 128, (unsigned long long)D);
        tdm_load_tile(ldsB0, gB, KSTEP, 128, (unsigned long long)D);
    }

    for (int s = 0; s < nstage; ++s) {
        const int cur = s & 1;
        if (wave == 0) {
            if (s + 1 < nstage) {          // issue next stage, then wait for cur
                unsigned lA = ((s + 1) & 1) ? ldsA1 : ldsA0;
                unsigned lB = ((s + 1) & 1) ? ldsB1 : ldsB0;
                int koff = (s + 1) * KSTEP;
                tdm_load_tile(lA, gA + koff, KSTEP, 128, (unsigned long long)D);
                tdm_load_tile(lB, gB + koff, KSTEP, 128, (unsigned long long)D);
                __builtin_amdgcn_s_wait_tensorcnt((short)2);
            } else {
                __builtin_amdgcn_s_wait_tensorcnt((short)0);
            }
        }
        __syncthreads();                   // stage `cur` resident in LDS

        const __bf16* pa = &sA[cur][0];
        const __bf16* pb = &sB[cur][0];

        // bf16 A(16x32) fragment: lane<16 holds K=[0..7]+[16..23], lane>=16
        // holds K=[8..15]+[24..31] of row M=lane&15 -> contiguous LDS runs.
        v16bf afrag[2], bfrag[4];
#pragma unroll
        for (int tm = 0; tm < 2; ++tm) {
            int rbase = (wm * 32 + tm * 16 + lrow) * LDS_STRIDE;
            v8bf lo = *(const v8bf*)(pa + rbase + half * 8);
            v8bf hi = *(const v8bf*)(pa + rbase + 16 + half * 8);
#pragma unroll
            for (int e = 0; e < 8; ++e) {
                afrag[tm][e]     = lo[e];
                afrag[tm][e + 8] = hi[e];
            }
        }
        // bf16 B(32x16) fragment: lane<16 holds K=[0..7]+[16..23], lane>=16
        // holds K=[8..15]+[24..31] of col N=lane&15.
#pragma unroll
        for (int tn = 0; tn < 4; ++tn) {
            int rbase = (wn * 64 + tn * 16 + lrow) * LDS_STRIDE;
            v8bf lo = *(const v8bf*)(pb + rbase + half * 8);
            v8bf hi = *(const v8bf*)(pb + rbase + 16 + half * 8);
#pragma unroll
            for (int e = 0; e < 8; ++e) {
                bfrag[tn][e]     = lo[e];
                bfrag[tn][e + 8] = hi[e];
            }
        }

#pragma unroll
        for (int tm = 0; tm < 2; ++tm)
#pragma unroll
            for (int tn = 0; tn < 4; ++tn)
                acc[tm][tn] = __builtin_amdgcn_wmma_f32_16x16x32_bf16(
                    false, afrag[tm], false, bfrag[tn],
                    (short)0, acc[tm][tn], false, false);

        __syncthreads();                   // done reading buffer `cur`
    }

    // ---------------- epilogue ----------------
    float colA[4] = {0.f, 0.f, 0.f, 0.f};
    float colB[4] = {0.f, 0.f, 0.f, 0.f};

#pragma unroll
    for (int tm = 0; tm < 2; ++tm) {
#pragma unroll
        for (int r = 0; r < 8; ++r) {
            // C/D layout: VGPR r, lanes 0-15 -> M=r, lanes 16-31 -> M=r+8; N=lane&15
            int m_loc  = wm * 32 + tm * 16 + half * 8 + r;
            float dm   = s_drow[m_loc];
            float em   = s_erow[m_loc];
            int   labm = s_lrow[m_loc];
            float rowA = 0.f, rowB = 0.f;
#pragma unroll
            for (int tn = 0; tn < 4; ++tn) {
                int n_loc = wn * 64 + tn * 16 + lrow;
                float c   = acc[tm][tn][r];
                float ecT = __expf(c * invT);
                int labn  = s_lcol[n_loc];
                if (labm == 1 && labn == 0) {          // pos row, neg col
                    float e = ecT * em;
                    rowA += e;
                    rowB += e * (c - dm);
                }
                if (labm == 0 && labn == 1) {          // neg row, pos col
                    float e = ecT * s_ecol[n_loc];
                    colA[tn] += e;
                    colB[tn] += e * (c - s_dcol[n_loc]);
                }
            }
            // lanes within a half share m -> 16-lane butterfly
#pragma unroll
            for (int msk = 8; msk >= 1; msk >>= 1) {
                rowA += __shfl_xor(rowA, msk, 16);
                rowB += __shfl_xor(rowB, msk, 16);
            }
            if (lrow == 0 && labm == 1) {
                atomicAdd(&Arow[row_base + m_loc], rowA);
                atomicAdd(&Brow[row_base + m_loc], rowB);
            }
        }
    }
#pragma unroll
    for (int tn = 0; tn < 4; ++tn) {
        float ca = colA[tn] + __shfl_xor(colA[tn], 16, 32);
        float cb = colB[tn] + __shfl_xor(colB[tn], 16, 32);
        int n_loc = wn * 64 + tn * 16 + lrow;
        if (half == 0 && s_lcol[n_loc] == 1) {
            atomicAdd(&Acol[col_base + n_loc], ca);
            atomicAdd(&Bcol[col_base + n_loc], cb);
        }
    }
}

// ---------------------------------------------------------------------------
// Per-sample CE over NUM_CT classes + segment sums (one wave per row).
// cls[0..7]=ce_sum, cls[8..15]=last_sum, cls[16..23]=count
// ---------------------------------------------------------------------------
__global__ __launch_bounds__(256)
void ce_kernel(const float* __restrict__ imgc, const float* __restrict__ txtc,
               const int* __restrict__ labels, const float* __restrict__ last,
               float* __restrict__ cls, int Bc, int D)
{
    int wave = threadIdx.x >> 5;
    int lane = threadIdx.x & 31;
    int row  = blockIdx.x * 8 + wave;
    if (row >= Bc) return;
    float dots[NUM_CT];
#pragma unroll
    for (int c = 0; c < NUM_CT; ++c) dots[c] = 0.f;
    const float* ip = imgc + (size_t)row * D;
    for (int k = lane; k < D; k += 32) {
        float a = ip[k];
#pragma unroll
        for (int c = 0; c < NUM_CT; ++c) dots[c] += a * txtc[c * D + k];
    }
#pragma unroll
    for (int c = 0; c < NUM_CT; ++c)
#pragma unroll
        for (int msk = 16; msk >= 1; msk >>= 1)
            dots[c] += __shfl_xor(dots[c], msk, 32);
    if (lane == 0) {
        float logit[NUM_CT], mx = -1e30f;
#pragma unroll
        for (int c = 0; c < NUM_CT; ++c) {
            logit[c] = dots[c] * (1.0f / TAU_C);
            mx = fmaxf(mx, logit[c]);
        }
        float se = 0.f;
#pragma unroll
        for (int c = 0; c < NUM_CT; ++c) se += __expf(logit[c] - mx);
        float lse = __logf(se) + mx;
        int lb = labels[row];
        float ce = lse - logit[lb];
        atomicAdd(&cls[lb], ce);
        atomicAdd(&cls[8 + lb], last[row]);
        atomicAdd(&cls[16 + lb], 1.0f);
    }
}

// ---------------------------------------------------------------------------
// Per-positive finalize: moving averages, scatter into out s_I/s_T copies,
// accumulate image/text loss sums.
// ---------------------------------------------------------------------------
__global__ __launch_bounds__(256)
void finalize_rows(const float* __restrict__ Arow, const float* __restrict__ Brow,
                   const float* __restrict__ Acol, const float* __restrict__ Bcol,
                   const int* __restrict__ slabel,
                   const int* __restrict__ image_ids, const int* __restrict__ text_ids,
                   const float* __restrict__ s_I, const float* __restrict__ s_T,
                   const int* __restrict__ epoch_p, const int* __restrict__ cnts,
                   float* __restrict__ out_sI, float* __restrict__ out_sT,
                   float* __restrict__ sums, int Bsz)
{
    int i = blockIdx.x * 256 + threadIdx.x;
    if (i >= Bsz) return;
    if (slabel[i] != 1) return;
    int epoch    = epoch_p[0];
    int cneg     = cnts[1];
    float inv_neg = 1.0f / (float)(cneg > 0 ? cneg : 1);

    float gI = Arow[i] * inv_neg;
    float sI = (epoch == 0) ? gI : ((1.0f - GAMMA1) * s_I[image_ids[i]] + GAMMA1 * gI);
    out_sI[image_ids[i]] = sI;
    atomicAdd(&sums[0], (Brow[i] * inv_neg) / (sI + EPS_F));

    float gT = Acol[i] * inv_neg;
    float sT = (epoch == 0) ? gT : ((1.0f - GAMMA1) * s_T[text_ids[i]] + GAMMA1 * gT);
    out_sT[text_ids[i]] = sT;
    atomicAdd(&sums[1], (Bcol[i] * inv_neg) / (sT + EPS_F));
}

// ---------------------------------------------------------------------------
// Final scalar: per-class means, u update, constraint, total loss.
// ---------------------------------------------------------------------------
__global__ void final_kernel(const float* __restrict__ cls, const float* __restrict__ sums,
                             const int* __restrict__ cnts, const float* __restrict__ u,
                             float* __restrict__ out, int N)
{
    if (threadIdx.x != 0 || blockIdx.x != 0) return;
    float mean_cls[NUM_CT], cv[NUM_CT];
    bool  pres[NUM_CT];
    float u_sum = 0.f;
#pragma unroll
    for (int c = 0; c < NUM_CT; ++c) {
        float cnt = cls[16 + c];
        pres[c] = cnt > 0.f;
        float safe = fmaxf(cnt, 1.f);
        mean_cls[c] = cls[c] / safe;
        cv[c] = mean_cls[c] - cls[8 + c] / safe;
        if (pres[c]) u_sum += u[c];
    }
    float csum = 0.f, n_present = 0.f;
#pragma unroll
    for (int c = 0; c < NUM_CT; ++c) {
        float uu = (u_sum == 0.f) ? cv[c] : ((1.0f - GAMMA2) * u[c] + GAMMA2 * cv[c]);
        float un = pres[c] ? uu : u[c];
        out[1 + 2 * (size_t)N + c] = un;
        if (pres[c]) {
            csum += fmaxf(BETA_C * un, 0.f) * mean_cls[c] * TAU_C;
            n_present += 1.f;
        }
    }
    int cp = cnts[0];
    float P = (float)(cp > 0 ? cp : 1);
    out[0] = sums[0] / P + sums[1] / P + csum / n_present;
}

// ---------------------------------------------------------------------------
extern "C" void kernel_launch(void* const* d_in, const int* in_sizes, int n_in,
                              void* d_out, int out_size, void* d_ws, size_t ws_size,
                              hipStream_t stream)
{
    const float* img       = (const float*)d_in[0];
    const float* txt       = (const float*)d_in[1];
    const int*   image_ids = (const int*)d_in[2];
    const int*   text_ids  = (const int*)d_in[3];
    const int*   slabel    = (const int*)d_in[4];
    const int*   epoch     = (const int*)d_in[5];
    const float* imgc      = (const float*)d_in[6];
    const float* txtc      = (const float*)d_in[7];
    const int*   labels_c  = (const int*)d_in[8];
    const float* last_loss = (const float*)d_in[9];
    const float* s_I       = (const float*)d_in[10];
    const float* s_T       = (const float*)d_in[11];
    const float* u         = (const float*)d_in[12];

    const int Bsz = in_sizes[2];            // 8192
    const int D   = in_sizes[0] / Bsz;      // 512
    const int Bc  = in_sizes[8];            // 4096
    const int N   = in_sizes[10];           // 1,000,000

    float* out = (float*)d_out;

    // workspace layout
    char*   ws   = (char*)d_ws;
    __bf16* imgB = (__bf16*)ws;
    __bf16* txtB = imgB + (size_t)Bsz * D;
    float*  diag = (float*)(txtB + (size_t)Bsz * D);
    float*  Arow = diag + Bsz;
    float*  Brow = Arow + Bsz;
    float*  Acol = Brow + Bsz;
    float*  Bcol = Acol + Bsz;
    float*  cls  = Bcol + Bsz;              // 24 floats
    float*  sums = cls + 24;                // 2 floats
    int*    cnts = (int*)(sums + 2);        // 2 ints

    // pass-through copies of persistent buffers (scattered updates come later)
    hipMemcpyAsync(out + 1, s_I, (size_t)N * sizeof(float), hipMemcpyDeviceToDevice, stream);
    hipMemcpyAsync(out + 1 + (size_t)N, s_T, (size_t)N * sizeof(float), hipMemcpyDeviceToDevice, stream);

    int nzero = 4 * Bsz + 24 + 2 + 2;
    zero_ws<<<(nzero + 255) / 256, 256, 0, stream>>>(Arow, nzero);

    convert_diag<<<Bsz, 256, 0, stream>>>(img, txt, imgB, txtB, diag, slabel, cnts, D);

    dim3 grid(Bsz / 128, Bsz / 128);
    gemm_epilogue<<<grid, 256, 0, stream>>>(imgB, txtB, diag, slabel,
                                            Arow, Brow, Acol, Bcol, D);

    ce_kernel<<<(Bc + 7) / 8, 256, 0, stream>>>(imgc, txtc, labels_c, last_loss, cls, Bc, D);

    finalize_rows<<<(Bsz + 255) / 256, 256, 0, stream>>>(
        Arow, Brow, Acol, Bcol, slabel, image_ids, text_ids,
        s_I, s_T, epoch, cnts, out + 1, out + 1 + (size_t)N, sums, Bsz);

    final_kernel<<<1, 32, 0, stream>>>(cls, sums, cnts, u, out, N);
}
